// DCNv2_8658654069328
// MI455X (gfx1250) — matrix operand order repaired and checked by reading, not compiled, not run on hardware
//
#include <hip/hip_runtime.h>
#include <hip/hip_bf16.h>

typedef __attribute__((ext_vector_type(16))) _Float16 v16h;
typedef __attribute__((ext_vector_type(8)))  _Float16 h8;
typedef __attribute__((ext_vector_type(8)))  float    v8f;

#define NB   4
#define HH_  64
#define WW_  64
#define CC   256
#define FF   256
#define KK9  9
#define NPIX (NB*HH_*WW_)          // 16384
#define MT   32                    // pixels per block (M tile)

// ---------------------------------------------------------------------------
// Kernel 1: offset conv  x(4,64,64,256) * offset_w(3,3,256,27) + b -> om(NPIX,27)
// ---------------------------------------------------------------------------
__global__ __launch_bounds__(256) void offset_conv_k(
    const float* __restrict__ x, const float* __restrict__ ow,
    const float* __restrict__ ob, float* __restrict__ om)
{
    const int tid = threadIdx.x;
    const int grp = tid >> 5;          // 8 pixels per block
    const int j   = tid & 31;          // output channel (27 active)
    const int pix = blockIdx.x * 8 + grp;
    const int n   = pix >> 12;
    const int h   = (pix >> 6) & 63;
    const int w   = pix & 63;
    const int jj  = (j < 27) ? j : 0;

    float acc = (j < 27) ? ob[jj] : 0.0f;
    #pragma unroll
    for (int kh = 0; kh < 3; ++kh) {
        const int y = h + kh - 1;
        if ((unsigned)y >= 64u) continue;
        #pragma unroll
        for (int kw = 0; kw < 3; ++kw) {
            const int xx = w + kw - 1;
            if ((unsigned)xx >= 64u) continue;
            const float* xp = x  + (((n * 64 + y) * 64 + xx) * CC);
            const float* wp = ow + ((kh * 3 + kw) * CC) * 27 + jj;
            #pragma unroll 4
            for (int c = 0; c < CC; ++c)
                acc = fmaf(xp[c], wp[c * 27], acc);
        }
    }
    if (j < 27) om[pix * 27 + j] = acc;
}

// ---------------------------------------------------------------------------
// Kernel 2: reorder dcn_weight (F,C,3,3) fp32 -> wB[tap][f][c] f16
//   w[K=c*9+t, f] = dcn_weight[f,c,t]  =>  wB[t*65536 + f*256 + c]
// ---------------------------------------------------------------------------
__global__ __launch_bounds__(256) void reorder_w_k(
    const float* __restrict__ dw, _Float16* __restrict__ wB)
{
    const int idx = blockIdx.x * 256 + threadIdx.x;
    if (idx >= KK9 * FF * CC) return;
    const int c = idx & 255;
    const int f = (idx >> 8) & 255;
    const int t = idx >> 16;
    wB[idx] = (_Float16)dw[(f * CC + c) * KK9 + t];
}

// ---------------------------------------------------------------------------
// Kernel 3: fused bilinear sampling + WMMA GEMM
//   block = 256 threads (8 wave32), owns 32 pixels x 256 out-channels
//   each wave: 2 M-subtiles x 2 N-subtiles (B fragments reused across M,
//   A fragments reused across N)
// ---------------------------------------------------------------------------
__global__ __launch_bounds__(256) void dcn_main_k(
    const float* __restrict__ x, const float* __restrict__ om,
    const _Float16* __restrict__ wB, float* __restrict__ out)
{
    __shared__ __align__(16) _Float16 lds_val[MT * CC];   // 16 KB A-tile (f16)

    const int tid    = threadIdx.x;
    const int wave   = tid >> 5;
    const int lane   = tid & 31;
    const int ln     = lane & 15;
    const int laneHi = lane >> 4;

    const int pixbase = blockIdx.x * MT;          // tile never crosses a row (64%32==0)
    const int n  = pixbase >> 12;
    const int h  = (pixbase >> 6) & 63;
    const int w0 = pixbase & 63;

    // sampling assignment: p = pixel in tile (0..31), 32 channels per thread
    const int p     = tid >> 3;
    const int cbase = (tid & 7) * 32;
    const int pix   = pixbase + p;
    const int wcol  = w0 + p;
    const float* xn = x + (size_t)n * 64 * 64 * CC;

    const int fbase = wave * 32;                  // each wave: 2 N-subtiles of 16
    v8f acc00 = {}; v8f acc01 = {};               // [m-subtile][n-subtile]
    v8f acc10 = {}; v8f acc11 = {};

    for (int t = 0; t < KK9; ++t) {
        // ---------------- phase 1: bilinear sample tap t into LDS -----------
        const int kh = t / 3, kw = t % 3;
        const float offy = om[pix * 27 + 2 * t];
        const float offx = om[pix * 27 + 2 * t + 1];
        const float mr   = om[pix * 27 + 18 + t];
        const float msk  = 1.0f / (1.0f + __expf(-mr));

        float py = (float)(h + kh)    + offy;     // base_y + inner_y + off
        float px = (float)(wcol + kw) + offx;
        py = fminf(fmaxf(py, 0.0f), 65.0f);
        px = fminf(fmaxf(px, 0.0f), 65.0f);
        const float y1f = floorf(py), x1f = floorf(px);
        const int y1 = (int)y1f, x1 = (int)x1f;
        const float ly = py - y1f, lx = px - x1f;
        const float hy = 1.0f - ly, hx = 1.0f - lx;
        const float w11 = hy * hx * msk, w12 = hy * lx * msk;
        const float w21 = ly * hx * msk, w22 = ly * lx * msk;

        #pragma unroll
        for (int half = 0; half < 2; ++half) {    // 2 x 16 channels / thread
            const int cb = cbase + half * 16;
            float4 v11[4], v12[4], v21[4], v22[4];
            #pragma unroll
            for (int q = 0; q < 2; ++q) {         // q: y1 / y1+1
                const int yy = y1 + q;
                const bool yok = (yy >= 1) && (yy <= 64);
                #pragma unroll
                for (int r = 0; r < 2; ++r) {     // r: x1 / x1+1
                    const int xx = x1 + r;
                    const bool ok = yok && (xx >= 1) && (xx <= 64);
                    const float4* src = (const float4*)(xn +
                        (size_t)(((yy - 1) * 64 + (xx - 1)) * CC + cb));
                    float4* dst = (q == 0) ? (r == 0 ? v11 : v12)
                                           : (r == 0 ? v21 : v22);
                    #pragma unroll
                    for (int i = 0; i < 4; ++i) {
                        float4 v = ok ? src[i] : make_float4(0.f, 0.f, 0.f, 0.f);
                        dst[i] = v;
                    }
                }
            }
            _Float16* ldst = &lds_val[p * CC + cb];
            #pragma unroll
            for (int i = 0; i < 4; ++i) {
                ldst[4 * i + 0] = (_Float16)(w11 * v11[i].x + w12 * v12[i].x + w21 * v21[i].x + w22 * v22[i].x);
                ldst[4 * i + 1] = (_Float16)(w11 * v11[i].y + w12 * v12[i].y + w21 * v21[i].y + w22 * v22[i].y);
                ldst[4 * i + 2] = (_Float16)(w11 * v11[i].z + w12 * v12[i].z + w21 * v21[i].z + w22 * v22[i].z);
                ldst[4 * i + 3] = (_Float16)(w11 * v11[i].w + w12 * v12[i].w + w21 * v21[i].w + w22 * v22[i].w);
            }
        }
        __syncthreads();

        // ---------------- phase 2: WMMA over K=256 channels of this tap -----
        const _Float16* wt = wB + (size_t)t * FF * CC;    // [f][c] f16
        #pragma unroll
        for (int kc = 0; kc < 8; ++kc) {                  // 32 channels per chunk
            // A fragments (16x32 f16) for both M-subtiles, from LDS
            v16h a0, a1;
            {
                const int ab0 = (ln) * CC + kc * 32;
                const h8 lo0 = *(const h8*)&lds_val[ab0 + 16 * laneHi];
                const h8 hi0 = *(const h8*)&lds_val[ab0 + 16 * laneHi + 8];
                const int ab1 = (16 + ln) * CC + kc * 32;
                const h8 lo1 = *(const h8*)&lds_val[ab1 + 16 * laneHi];
                const h8 hi1 = *(const h8*)&lds_val[ab1 + 16 * laneHi + 8];
                #pragma unroll
                for (int i = 0; i < 8; ++i) {
                    a0[i] = lo0[i]; a0[i + 8] = hi0[i];
                    a1[i] = lo1[i]; a1[i + 8] = hi1[i];
                }
            }
            // B fragments (32x16 f16): lane holds 16 contiguous K (=c) values
            const v16h b0 = *(const v16h*)&wt[(fbase + ln)      * CC + kc * 32 + 16 * laneHi];
            const v16h b1 = *(const v16h*)&wt[(fbase + 16 + ln) * CC + kc * 32 + 16 * laneHi];

            acc00 = __builtin_amdgcn_wmma_f32_16x16x32_f16(
                        false, a0, false, b0, (short)0, acc00, false, false);
            acc01 = __builtin_amdgcn_wmma_f32_16x16x32_f16(
                        false, a0, false, b1, (short)0, acc01, false, false);
            acc10 = __builtin_amdgcn_wmma_f32_16x16x32_f16(
                        false, a1, false, b0, (short)0, acc10, false, false);
            acc11 = __builtin_amdgcn_wmma_f32_16x16x32_f16(
                        false, a1, false, b1, (short)0, acc11, false, false);
        }
        __syncthreads();   // protect LDS before next tap overwrites it
    }

    // ---------------- epilogue: C/D layout -> out(N,H,W,F) fp32 -------------
    #pragma unroll
    for (int r = 0; r < 8; ++r) {
        const int mrow = r + 8 * laneHi;
        float* op0 = out + (size_t)(pixbase + mrow) * FF + fbase;
        float* op1 = out + (size_t)(pixbase + 16 + mrow) * FF + fbase;
        op0[ln]      = acc00[r];
        op0[16 + ln] = acc01[r];
        op1[ln]      = acc10[r];
        op1[16 + ln] = acc11[r];
    }
}

// ---------------------------------------------------------------------------
extern "C" void kernel_launch(void* const* d_in, const int* in_sizes, int n_in,
                              void* d_out, int out_size, void* d_ws, size_t ws_size,
                              hipStream_t stream)
{
    const float* x    = (const float*)d_in[0];   // (4,64,64,256)
    const float* ow   = (const float*)d_in[1];   // (3,3,256,27)
    const float* ob   = (const float*)d_in[2];   // (27)
    const float* dw   = (const float*)d_in[3];   // (256,256,3,3)
    float*       out  = (float*)d_out;           // (4,64,64,256)

    // workspace layout
    float*     om = (float*)d_ws;                              // NPIX*27 fp32 = 1.77 MB
    _Float16*  wB = (_Float16*)((char*)d_ws + (size_t)NPIX * 27 * sizeof(float)); // 1.18 MB

    // 1) offset conv: 8 pixels per block
    offset_conv_k<<<NPIX / 8, 256, 0, stream>>>(x, ow, ob, om);
    // 2) weight reorder to f16 [tap][f][c]
    reorder_w_k<<<(KK9 * FF * CC + 255) / 256, 256, 0, stream>>>(dw, wB);
    // 3) fused sampling + WMMA GEMM: one 32-pixel tile per block
    dcn_main_k<<<NPIX / MT, 256, 0, stream>>>(x, om, wB, out);
}
